// DefTet_71571335020832
// MI455X (gfx1250) — compile-verified, matrix operand order checked by reading, not run on hardware
//
#include <hip/hip_runtime.h>
#include <hip/hip_bf16.h>

typedef unsigned short u16;
typedef unsigned int   u32;
typedef __attribute__((ext_vector_type(16))) __bf16 v16bf;
typedef __attribute__((ext_vector_type(8)))  float  v8f;
typedef __attribute__((ext_vector_type(4)))  u32    u32x4;

// ---------- helpers ----------
__device__ __forceinline__ u16 f2bf(float f) {
  u32 u = __builtin_bit_cast(u32, f);
  u32 r = (u + 0x7FFFu + ((u >> 16) & 1u)) >> 16;
  return (u16)r;
}
__device__ __forceinline__ float bf2f(u16 h) {
  u32 u = ((u32)h) << 16;
  return __builtin_bit_cast(float, u);
}

union FragU { u32x4 q[2]; v16bf v; };

__device__ __forceinline__ v8f wmma_bf16(v16bf a, v16bf b, v8f c) {
  // D(f32,16x16) = A(bf16,16x32) * B(bf16,32x16) + C
  return __builtin_amdgcn_wmma_f32_16x16x32_bf16(false, a, false, b, (short)0, c,
                                                 false, false);
}

// gfx1250 async global->LDS copy (per-lane 16B), tracked by ASYNCcnt.
// %0 = VGPR with LDS byte offset (dynamic-LDS base assumed 0 in these kernels),
// %1 = VGPR pair with 64-bit global address.
__device__ __forceinline__ void async_g2l_b128(u32 ldsByteOff, const void* gptr) {
  asm volatile("global_load_async_to_lds_b128 %0, %1, off"
               :: "v"(ldsByteOff), "v"(gptr) : "memory");
}
__device__ __forceinline__ void wait_async0() {
  asm volatile("s_wait_asynccnt 0x0" ::: "memory");
}

// ---------- elementwise ----------
__global__ void cvt_bf16(const float* __restrict__ x, u16* __restrict__ y, long n) {
  for (long i = (long)blockIdx.x * blockDim.x + threadIdx.x; i < n;
       i += (long)gridDim.x * blockDim.x)
    y[i] = f2bf(x[i]);
}

__global__ void zero_f32(float* __restrict__ p, long n) {
  for (long i = (long)blockIdx.x * blockDim.x + threadIdx.x; i < n;
       i += (long)gridDim.x * blockDim.x)
    p[i] = 0.f;
}

// ---------- weight repack into per-lane WMMA B fragments ----------
// frag element e (0..15) of lane L holds B[k][n] with
//   g=e>>1, h=e&1, klocal = ((g&4)<<2)|((L>>4)<<3)|((g&3)<<1)|h, n = ntile*16+(L&15)
// transposed=1: w is [N][K] (conv OIDHW flattened); transposed=0: w is [K][N]
__global__ void repack_w(const float* __restrict__ w, u16* __restrict__ frags,
                         int Kreal, int Nreal, int Ktiles, int NtilesTotal,
                         int ntOff, int NtilesLocal, int transposed) {
  long total = (long)Ktiles * NtilesLocal * 512;
  for (long i = (long)blockIdx.x * blockDim.x + threadIdx.x; i < total;
       i += (long)gridDim.x * blockDim.x) {
    int e = (int)(i & 15);
    int lane = (int)((i >> 4) & 31);
    long t = i >> 9;
    int ntl = (int)(t % NtilesLocal);
    int kt  = (int)(t / NtilesLocal);
    int g = e >> 1, h = e & 1;
    int klocal = ((g & 4) << 2) | ((lane >> 4) << 3) | ((g & 3) << 1) | h;
    int k  = kt * 32 + klocal;
    int nw = ntl * 16 + (lane & 15);
    float v = 0.f;
    if (k < Kreal && nw < Nreal)
      v = transposed ? w[(size_t)nw * Kreal + k] : w[(size_t)k * Nreal + nw];
    frags[(((size_t)kt * NtilesTotal + (ntOff + ntl)) * 32 + lane) * 16 + e] = f2bf(v);
  }
}

// ---------- implicit-GEMM conv3d with WMMA (all dims compile-time) ----------
// act: bf16 [nB][Cin][D][H][W]; Bf: repacked frags; out: f32 [nB][Cout][S]
// Block = 8 waves; M-tile 64 (4 sub-tiles of 16), each wave owns one 16-wide N-tile.
template <int nB, int Cin, int inDim, int outDim, int stride, int pad, int ksz,
          int Cout, int Kreal>
__global__ __launch_bounds__(256) void conv3d_wmma(const u16* __restrict__ act,
                                                   const u16* __restrict__ Bf,
                                                   float* __restrict__ out) {
  constexpr int Ntiles = Cout / 16;
  constexpr int Ktiles = (Kreal + 31) / 32;
  constexpr int outS = outDim * outDim * outDim;
  constexpr int khw = ksz * ksz;
  constexpr int kvol = khw * ksz;
  constexpr int inHW = inDim * inDim;

  __shared__ __align__(16) u16 As[64 * 32];
  const int lane = (int)threadIdx.x & 31;
  const int wave = (int)threadIdx.x >> 5;
  const int mtile = blockIdx.x;                 // 64-row M tile
  const int ntile = blockIdx.y * 8 + wave;
  const bool active = (ntile < Ntiles);

  v8f z = {0.f, 0.f, 0.f, 0.f, 0.f, 0.f, 0.f, 0.f};
  v8f acc[4] = {z, z, z, z};
  const int arow  = lane & 15;
  const int kbase = (lane >> 4) << 3;

  for (int kt = 0; kt < Ktiles; ++kt) {
    __syncthreads();
    // stage 64x32 A tile via on-the-fly im2col: 2048 elems / 256 threads
#pragma unroll
    for (int j = 0; j < 8; ++j) {
      int e = (int)threadIdx.x + j * 256;
      int r = e >> 5;
      int kk = e & 31;
      int k = kt * 32 + kk;
      int m = mtile * 64 + r;
      int b = m / outS;
      int s = m - b * outS;
      int oz = s / (outDim * outDim);
      int t2 = s - oz * outDim * outDim;
      int oy = t2 / outDim;
      int ox = t2 - oy * outDim;
      u16 v = 0;
      if (k < Kreal) {
        int ci = k / kvol;
        int kr = k - ci * kvol;
        int kz = kr / khw;
        int kr2 = kr - kz * khw;
        int ky = kr2 / ksz;
        int kx = kr2 - ky * ksz;
        int iz = oz * stride - pad + kz;
        int iy = oy * stride - pad + ky;
        int ix = ox * stride - pad + kx;
        if (iz >= 0 && iz < inDim && iy >= 0 && iy < inDim && ix >= 0 && ix < inDim)
          v = act[(size_t)(b * Cin + ci) * (size_t)(inDim * inHW) +
                  (size_t)iz * inHW + (size_t)iy * inDim + ix];
      }
      As[r * 32 + kk] = v;
    }
    __syncthreads();
    if (active) {
      FragU bfr;
      const u32x4* bp =
          (const u32x4*)(Bf + ((((size_t)kt * Ntiles + ntile) * 32 + lane) << 4));
      bfr.q[0] = bp[0];
      bfr.q[1] = bp[1];
      if (kt + 1 < Ktiles)
        __builtin_prefetch(
            Bf + ((((size_t)(kt + 1) * Ntiles + ntile) * 32 + lane) << 4), 0, 1);
#pragma unroll
      for (int mi = 0; mi < 4; ++mi) {
        FragU a;
        a.q[0] = *(const u32x4*)(As + (mi * 16 + arow) * 32 + kbase);
        a.q[1] = *(const u32x4*)(As + (mi * 16 + arow) * 32 + kbase + 16);
        acc[mi] = wmma_bf16(a.v, bfr.v, acc[mi]);
      }
    }
  }

  if (active) {
    int n = ntile * 16 + (lane & 15);
#pragma unroll
    for (int mi = 0; mi < 4; ++mi) {
      int mbase = mtile * 64 + mi * 16 + ((lane >> 4) << 3);
#pragma unroll
      for (int r = 0; r < 8; ++r) {
        int m = mbase + r;
        int b = m / outS;
        int s = m - b * outS;
        out[(size_t)(b * Cout + n) * outS + s] = acc[mi][r];
      }
    }
  }
}

// ---------- dense GEMM with WMMA + async global->LDS A staging ----------
// A: bf16 [M][Kpad] row-major; out f32 or bf16 with row stride Nout.
// Dynamic LDS (4 KB): 64x32 bf16 A tile; no static LDS => dyn base offset 0.
template <int Kpad, int Ntiles, int Nout, int outBf16>
__global__ __launch_bounds__(256) void gemm_wmma(const u16* __restrict__ A,
                                                 const u16* __restrict__ Bf,
                                                 void* __restrict__ Cout) {
  extern __shared__ __align__(16) u16 As[];
  constexpr int Ktiles = Kpad / 32;
  const int lane = (int)threadIdx.x & 31;
  const int wave = (int)threadIdx.x >> 5;
  const int mtile = blockIdx.x;                 // 64-row M tile
  const int ntile = blockIdx.y * 8 + wave;
  const bool active = (ntile < Ntiles);

  v8f z = {0.f, 0.f, 0.f, 0.f, 0.f, 0.f, 0.f, 0.f};
  v8f acc[4] = {z, z, z, z};
  const int arow  = lane & 15;
  const int kbase = (lane >> 4) << 3;

  // per-thread async-copy coordinates: thread t copies 16B of row (t>>2)
  const int crow = (int)threadIdx.x >> 2;       // 0..63
  const int cseg = (int)threadIdx.x & 3;        // 4 x 8 u16 segments
  const u32 ldsOff = (u32)((crow * 32 + cseg * 8) * 2);
  const u16* gbase = A + (size_t)(mtile * 64 + crow) * Kpad + cseg * 8;

  for (int kt = 0; kt < Ktiles; ++kt) {
    __syncthreads();                            // previous tile fully consumed
    async_g2l_b128(ldsOff, gbase + kt * 32);    // ASYNCcnt++
    wait_async0();                              // s_wait_asynccnt 0
    __syncthreads();                            // tile visible to all waves
    if (active) {
      FragU bfr;
      const u32x4* bp =
          (const u32x4*)(Bf + ((((size_t)kt * Ntiles + ntile) * 32 + lane) << 4));
      bfr.q[0] = bp[0];
      bfr.q[1] = bp[1];
      if (kt + 1 < Ktiles)
        __builtin_prefetch(
            Bf + ((((size_t)(kt + 1) * Ntiles + ntile) * 32 + lane) << 4), 0, 1);
#pragma unroll
      for (int mi = 0; mi < 4; ++mi) {
        FragU a;
        a.q[0] = *(const u32x4*)(As + (mi * 16 + arow) * 32 + kbase);
        a.q[1] = *(const u32x4*)(As + (mi * 16 + arow) * 32 + kbase + 16);
        acc[mi] = wmma_bf16(a.v, bfr.v, acc[mi]);
      }
    }
  }

  if (active) {
    int n = ntile * 16 + (lane & 15);
#pragma unroll
    for (int mi = 0; mi < 4; ++mi) {
      size_t mbase = (size_t)mtile * 64 + mi * 16 + ((lane >> 4) << 3);
      if (outBf16) {
        u16* o = (u16*)Cout;
#pragma unroll
        for (int r = 0; r < 8; ++r) o[(mbase + r) * Nout + n] = f2bf(acc[mi][r]);
      } else {
        float* o = (float*)Cout;
#pragma unroll
        for (int r = 0; r < 8; ++r) o[(mbase + r) * Nout + n] = acc[mi][r];
      }
    }
  }
}

// ---------- batchnorm ----------
__global__ void bn_stats(const float* __restrict__ x, float* __restrict__ scale,
                         float* __restrict__ shift, const float* __restrict__ g,
                         const float* __restrict__ bia, int C, int S, int nB) {
  int c = blockIdx.x;
  float sum = 0.f, sq = 0.f;
  long total = (long)nB * S;
  for (long i = threadIdx.x; i < total; i += blockDim.x) {
    int b = (int)(i / S);
    int s = (int)(i - (long)b * S);
    float v = x[(size_t)(b * C + c) * S + s];
    sum += v;
    sq += v * v;
  }
  __shared__ float s1[256], s2[256];
  s1[threadIdx.x] = sum;
  s2[threadIdx.x] = sq;
  __syncthreads();
  for (int st = 128; st > 0; st >>= 1) {
    if ((int)threadIdx.x < st) {
      s1[threadIdx.x] += s1[threadIdx.x + st];
      s2[threadIdx.x] += s2[threadIdx.x + st];
    }
    __syncthreads();
  }
  if (threadIdx.x == 0) {
    float inv = 1.f / (float)total;
    float m = s1[0] * inv;
    float v = s2[0] * inv - m * m;
    float rs = rsqrtf(v + 1e-5f);
    float sc = g[c] * rs;
    scale[c] = sc;
    shift[c] = bia[c] - m * sc;
  }
}

__global__ void bn_apply(const float* __restrict__ x, const float* __restrict__ scale,
                         const float* __restrict__ shift, u16* __restrict__ y,
                         int C, int S, long total, int doRelu) {
  for (long i = (long)blockIdx.x * blockDim.x + threadIdx.x; i < total;
       i += (long)gridDim.x * blockDim.x) {
    int c = (int)((i / S) % C);
    float v = x[i] * scale[c] + shift[c];
    if (doRelu) v = fmaxf(v, 0.f);
    y[i] = f2bf(v);
  }
}

// ---------- grid -> graph gather + concat (rows padded 515 -> 544) ----------
__global__ void gather_nodes(const float* __restrict__ pos, const u16* __restrict__ f2,
                             u16* __restrict__ nodes, long total) {
  for (long i = (long)blockIdx.x * blockDim.x + threadIdx.x; i < total;
       i += (long)gridDim.x * blockDim.x) {
    long row = i / 544;
    int c = (int)(i - row * 544);
    int b = (int)(row / 50000);
    int nd = (int)(row - (long)b * 50000);
    u16 v = 0;
    if (c < 3) {
      v = f2bf(pos[nd * 3 + c]);
    } else if (c < 515) {
      int ch = c - 3;
      int iz = (int)(pos[nd * 3 + 0] * 16.f - 1.f);
      int iy = (int)(pos[nd * 3 + 1] * 16.f - 1.f);
      int ix = (int)(pos[nd * 3 + 2] * 16.f - 1.f);
      v = f2[(((size_t)(b * 512 + ch) * 16 + iz) * 16 + iy) * 16 + ix];
    }
    nodes[i] = v;
  }
}

// ---------- gated edge message + segment_sum (atomic f32) ----------
// kqvs: bf16 [rows][Cstride] laid out as [k | q | v | s] blocks of C
__global__ void edge_gate(const int* __restrict__ ei, const u16* __restrict__ kqvs,
                          float* __restrict__ agg, int E, int nNodes, int C,
                          int Cstride) {
  int eb = blockIdx.x;
  int b = eb / E;
  int i = eb - b * E;
  int s = ei[i] + b * nNodes;
  int d = ei[E + i] + b * nNodes;
  int c = threadIdx.x;
  float kd = bf2f(kqvs[(size_t)d * Cstride + c]);
  float qs = bf2f(kqvs[(size_t)s * Cstride + C + c]);
  float vs = bf2f(kqvs[(size_t)s * Cstride + 2 * C + c]);
  float gate = 1.f / (1.f + __expf(-(kd + qs)));
  (void)__hip_atomic_fetch_add(&agg[(size_t)d * C + c], gate * vs, __ATOMIC_RELAXED,
                               __HIP_MEMORY_SCOPE_AGENT);
}

// h = relu(agg + x@Ws + bias)  (x@Ws is the 4th block of kqvs)
__global__ void gate_combine(const float* __restrict__ agg, const u16* __restrict__ kqvs,
                             const float* __restrict__ bias, u16* __restrict__ hout,
                             int C, int Cstride, long total) {
  for (long i = (long)blockIdx.x * blockDim.x + threadIdx.x; i < total;
       i += (long)gridDim.x * blockDim.x) {
    long r = i / C;
    int c = (int)(i - r * C);
    float v = agg[i] + bf2f(kqvs[(size_t)r * Cstride + 3 * C + c]) + bias[c];
    hout[i] = f2bf(fmaxf(v, 0.f));
  }
}

__global__ void bias_relu_bf16(const float* __restrict__ x, const float* __restrict__ bias,
                               u16* __restrict__ y, int C, long total) {
  for (long i = (long)blockIdx.x * blockDim.x + threadIdx.x; i < total;
       i += (long)gridDim.x * blockDim.x) {
    int c = (int)(i % C);
    y[i] = f2bf(fmaxf(x[i] + bias[c], 0.f));
  }
}

// final: relu(g4[r][0..2] + m2b) -> d_out [rows][3] f32 (g4 row stride 16)
__global__ void final_out(const float* __restrict__ x16, const float* __restrict__ bias,
                          float* __restrict__ out, long total) {
  for (long i = (long)blockIdx.x * blockDim.x + threadIdx.x; i < total;
       i += (long)gridDim.x * blockDim.x) {
    long r = i / 3;
    int c = (int)(i - r * 3);
    out[i] = fmaxf(x16[r * 16 + c] + bias[c], 0.f);
  }
}

// ---------- host ----------
extern "C" void kernel_launch(void* const* d_in, const int* in_sizes, int n_in,
                              void* d_out, int out_size, void* d_ws, size_t ws_size,
                              hipStream_t stream) {
  (void)in_sizes; (void)n_in; (void)out_size; (void)ws_size;
  const float* x   = (const float*)d_in[0];
  const float* pos = (const float*)d_in[1];
  const float* w0  = (const float*)d_in[2];
  const float* g0  = (const float*)d_in[3];
  const float* b0  = (const float*)d_in[4];
  const float* w1  = (const float*)d_in[5];
  const float* g1  = (const float*)d_in[6];
  const float* b1  = (const float*)d_in[7];
  const float* w2  = (const float*)d_in[8];
  const float* g2  = (const float*)d_in[9];
  const float* b2  = (const float*)d_in[10];
  const float* gk1 = (const float*)d_in[11];
  const float* gq1 = (const float*)d_in[12];
  const float* gv1 = (const float*)d_in[13];
  const float* gs1 = (const float*)d_in[14];
  const float* gb1 = (const float*)d_in[15];
  const float* gk2 = (const float*)d_in[16];
  const float* gq2 = (const float*)d_in[17];
  const float* gv2 = (const float*)d_in[18];
  const float* gs2 = (const float*)d_in[19];
  const float* gb2 = (const float*)d_in[20];
  const float* m1w = (const float*)d_in[21];
  const float* m1b = (const float*)d_in[22];
  const float* m2w = (const float*)d_in[23];
  const float* m2b = (const float*)d_in[24];
  const int*   ei  = (const int*)d_in[25];
  float* outp = (float*)d_out;

  char* ws = (char*)d_ws;
  const size_t MB = 1024ull * 1024ull;
  float* bnScale = (float*)(ws);                 // 4 KB
  float* bnShift = (float*)(ws + 64 * 1024);
  float* bigF    = (float*)(ws + 1 * MB);        // conv raw f32 / agg (100 MB)
  u16*   actD    = (u16*)(ws + 101 * MB);        // act0 / h1 / h2 / h3 (52 MB)
  u16*   wfrag   = (u16*)(ws + 153 * MB);        // weight fragments (17 MB)
  u16*   act1b   = (u16*)(ws + 170 * MB);        // act1 (9 MB)
  u16*   f2buf   = (u16*)(ws + 179 * MB);        // f2 bf16 (9 MB)
  u16*   bufH    = (u16*)(ws + 188 * MB);        // nodes / kqvs2 / mlp outs (110 MB)
  u16*   kqvs1   = (u16*)(ws + 298 * MB);        // layer-1 k|q|v|s bf16 (205 MB)
  u16*   xbf     = (u16*)(ws + 503 * MB);        // x in bf16 (1 MB)
  float* mlpF    = (float*)bufH;                 // fp32 mlp scratch (after kqvs2 dead)

  dim3 blk(256);
  const size_t lds = 64 * 32 * sizeof(u16);      // dynamic LDS for gemm_wmma

  // ---- conv stack ----
  cvt_bf16<<<dim3(512), blk, 0, stream>>>(x, xbf, 262144L);

  repack_w<<<dim3(256), blk, 0, stream>>>(w0, wfrag, 108, 256, 4, 16, 0, 16, 1);
  conv3d_wmma<2, 4, 32, 32, 1, 1, 3, 256, 108>
      <<<dim3(1024, 2), blk, 0, stream>>>(xbf, wfrag, bigF);
  bn_stats<<<dim3(256), blk, 0, stream>>>(bigF, bnScale, bnShift, g0, b0, 256, 32768, 2);
  bn_apply<<<dim3(2048), blk, 0, stream>>>(bigF, bnScale, bnShift, actD,
      256, 32768, 16777216L, 1);

  repack_w<<<dim3(8192), blk, 0, stream>>>(w1, wfrag, 16384, 512, 512, 32, 0, 32, 1);
  conv3d_wmma<2, 256, 32, 16, 2, 1, 4, 512, 16384>
      <<<dim3(128, 4), blk, 0, stream>>>(actD, wfrag, bigF);
  bn_stats<<<dim3(512), blk, 0, stream>>>(bigF, bnScale, bnShift, g1, b1, 512, 4096, 2);
  bn_apply<<<dim3(2048), blk, 0, stream>>>(bigF, bnScale, bnShift, act1b,
      512, 4096, 4194304L, 1);

  repack_w<<<dim3(8192), blk, 0, stream>>>(w2, wfrag, 13824, 512, 432, 32, 0, 32, 1);
  conv3d_wmma<2, 512, 16, 16, 1, 1, 3, 512, 13824>
      <<<dim3(128, 4), blk, 0, stream>>>(act1b, wfrag, bigF);
  bn_stats<<<dim3(512), blk, 0, stream>>>(bigF, bnScale, bnShift, g2, b2, 512, 4096, 2);
  bn_apply<<<dim3(2048), blk, 0, stream>>>(bigF, bnScale, bnShift, f2buf,
      512, 4096, 4194304L, 0);  // no relu (gather source)

  // ---- build node features [100000][544] bf16 ----
  gather_nodes<<<dim3(4096), blk, 0, stream>>>(pos, f2buf, bufH, 100000L * 544L);

  // ---- GNN layer 1: fused K|Q|V|S GEMM (M=100000 -> 1563x64, K=544, N=1024) ----
  repack_w<<<dim3(1024), blk, 0, stream>>>(gk1, wfrag, 515, 256, 17, 64, 0,  16, 0);
  repack_w<<<dim3(1024), blk, 0, stream>>>(gq1, wfrag, 515, 256, 17, 64, 16, 16, 0);
  repack_w<<<dim3(1024), blk, 0, stream>>>(gv1, wfrag, 515, 256, 17, 64, 32, 16, 0);
  repack_w<<<dim3(1024), blk, 0, stream>>>(gs1, wfrag, 515, 256, 17, 64, 48, 16, 0);
  gemm_wmma<544, 64, 1024, 1>
      <<<dim3(1563, 8), blk, lds, stream>>>(bufH, wfrag, kqvs1);

  zero_f32<<<dim3(2048), blk, 0, stream>>>(bigF, 25600000L);
  edge_gate<<<dim3(600000), dim3(256), 0, stream>>>(ei, kqvs1, bigF, 300000, 50000,
                                                    256, 1024);
  gate_combine<<<dim3(2048), blk, 0, stream>>>(bigF, kqvs1, gb1, actD,
                                               256, 1024, 25600000L);

  // ---- GNN layer 2: (M=100000, K=256, N=512) ----
  repack_w<<<dim3(512), blk, 0, stream>>>(gk2, wfrag, 256, 128, 8, 32, 0,  8, 0);
  repack_w<<<dim3(512), blk, 0, stream>>>(gq2, wfrag, 256, 128, 8, 32, 8,  8, 0);
  repack_w<<<dim3(512), blk, 0, stream>>>(gv2, wfrag, 256, 128, 8, 32, 16, 8, 0);
  repack_w<<<dim3(512), blk, 0, stream>>>(gs2, wfrag, 256, 128, 8, 32, 24, 8, 0);
  gemm_wmma<256, 32, 512, 1>
      <<<dim3(1563, 4), blk, lds, stream>>>(actD, wfrag, bufH);

  zero_f32<<<dim3(2048), blk, 0, stream>>>(bigF, 12800000L);
  edge_gate<<<dim3(600000), dim3(128), 0, stream>>>(ei, bufH, bigF, 300000, 50000,
                                                    128, 512);
  gate_combine<<<dim3(2048), blk, 0, stream>>>(bigF, bufH, gb2, actD,
                                               128, 512, 12800000L);

  // ---- MLP head ----
  repack_w<<<dim3(32), blk, 0, stream>>>(m1w, wfrag, 128, 64, 4, 4, 0, 4, 0);
  gemm_wmma<128, 4, 64, 0>
      <<<dim3(1563, 1), blk, lds, stream>>>(actD, wfrag, (void*)mlpF);
  bias_relu_bf16<<<dim3(2048), blk, 0, stream>>>(mlpF, m1b, actD, 64, 6400000L);

  repack_w<<<dim3(8), blk, 0, stream>>>(m2w, wfrag, 64, 3, 2, 1, 0, 1, 0);
  gemm_wmma<64, 1, 16, 0>
      <<<dim3(1563, 1), blk, lds, stream>>>(actD, wfrag, (void*)mlpF);
  final_out<<<dim3(1024), blk, 0, stream>>>(mlpF, m2b, outp, 300000L);
}
